// HCSFEngine_81509889343911
// MI455X (gfx1250) — matrix operand order repaired and checked by reading, not compiled, not run on hardware
//
#include <hip/hip_runtime.h>
#include <math.h>

#define B_      4
#define L_      2048
#define D_      512
#define HID_    1024
#define TOPK_   4
#define KSTEPS_ 10
#define CHAIN_  (L_ - 1)             // 2047
#define E_      (CHAIN_ + TOPK_*L_)  // 10239
#define M_      (B_*E_*2)            // 81912
#define LAM_    0.01f
#define EPSC_   1e-7f
#define MAXGN_  1.0f
#define MAXREL_ 1024
#define NEGINF_ (-3.402823466e+38f)

typedef __bf16 bf16_t;
typedef bf16_t v16bf __attribute__((ext_vector_type(16)));
typedef float  v8f   __attribute__((ext_vector_type(8)));
typedef unsigned int u32x4 __attribute__((ext_vector_type(4)));
typedef int          i32x4 __attribute__((ext_vector_type(4)));
typedef int          i32x8 __attribute__((ext_vector_type(8)));

static __device__ __forceinline__ unsigned short f2bf(float f) {
    unsigned int u = __float_as_uint(f);
    unsigned int r = u + 0x7FFFu + ((u >> 16) & 1u);   // round-to-nearest-even
    return (unsigned short)(r >> 16);
}

// ---------------------------------------------------------------- top-k ----
static __device__ __forceinline__ void top4_insert(float* v, int* id, float x, int j) {
    if (x <= v[3]) return;
    v[3] = x; id[3] = j;
#pragma unroll
    for (int s = 3; s > 0; --s) {
        if (v[s] > v[s-1]) {
            float tv = v[s]; v[s] = v[s-1]; v[s-1] = tv;
            int   ti = id[s]; id[s] = id[s-1]; id[s-1] = ti;
        }
    }
}

__global__ void k_topk(const float* __restrict__ attn,
                       float* __restrict__ vals, int* __restrict__ idxs) {
    int b = blockIdx.x / L_;
    int i = blockIdx.x % L_;
    int lane = threadIdx.x;                 // block = 32 (one wave)
    const float* row = attn + ((size_t)(b*L_ + i)) * L_;
    float v[TOPK_]; int id[TOPK_];
#pragma unroll
    for (int s = 0; s < TOPK_; ++s) { v[s] = NEGINF_; id[s] = s; }
    for (int j = lane; j < i; j += 32) top4_insert(v, id, row[j], j);
    // butterfly merge across the wave
    for (int off = 16; off; off >>= 1) {
        float ov[TOPK_]; int oi[TOPK_];
#pragma unroll
        for (int s = 0; s < TOPK_; ++s) {
            ov[s] = __shfl_xor(v[s], off, 32);
            oi[s] = __shfl_xor(id[s], off, 32);
        }
#pragma unroll
        for (int s = 0; s < TOPK_; ++s) top4_insert(v, id, ov[s], oi[s]);
    }
    if (lane == 0) {
        size_t base = ((size_t)(b*L_ + i)) * TOPK_;
#pragma unroll
        for (int s = 0; s < TOPK_; ++s) { vals[base+s] = v[s]; idxs[base+s] = id[s]; }
    }
}

// softmax over axis=1 (rows) per (b, k); writes the top-k edges
__global__ void k_colsm(const float* __restrict__ vals, const int* __restrict__ idxs,
                        int* esrc, int* etgt, int* erel, float* ew) {
    __shared__ float red[256];
    int b = blockIdx.x / TOPK_;
    int k = blockIdx.x % TOPK_;
    int tid = threadIdx.x;
    float mx = NEGINF_;
    for (int i = tid; i < L_; i += 256)
        mx = fmaxf(mx, vals[((size_t)(b*L_ + i))*TOPK_ + k]);
    red[tid] = mx; __syncthreads();
    for (int s = 128; s; s >>= 1) { if (tid < s) red[tid] = fmaxf(red[tid], red[tid+s]); __syncthreads(); }
    mx = red[0]; __syncthreads();
    float sm = 0.f;
    for (int i = tid; i < L_; i += 256) {
        float x = vals[((size_t)(b*L_ + i))*TOPK_ + k];
        if (x > NEGINF_) sm += expf(x - mx);
    }
    red[tid] = sm; __syncthreads();
    for (int s = 128; s; s >>= 1) { if (tid < s) red[tid] += red[tid+s]; __syncthreads(); }
    float inv = 1.0f / red[0];
    for (int i = tid; i < L_; i += 256) {
        size_t vi = ((size_t)(b*L_ + i))*TOPK_ + k;
        float x = vals[vi];
        int   j = idxs[vi];
        float w = (x > NEGINF_) ? expf(x - mx) * inv : 0.0f;
        size_t be = (size_t)b*E_ + (CHAIN_ + k*L_ + i);
        esrc[be] = i; etgt[be] = j; erel[be] = j - i; ew[be] = w;
    }
}

__global__ void k_chain(const float* __restrict__ attn,
                        int* esrc, int* etgt, int* erel, float* ew) {
    int t = blockIdx.x * blockDim.x + threadIdx.x;
    if (t >= B_ * CHAIN_) return;
    int b = t / CHAIN_, i = t % CHAIN_;
    size_t be = (size_t)b*E_ + i;
    esrc[be] = i + 1; etgt[be] = i; erel[be] = -1;
    ew[be] = attn[((size_t)(b*L_ + i + 1))*L_ + i];
}

// ------------------------------------------------------------- build X ----
__global__ void k_buildX(const float* __restrict__ h, const float* __restrict__ pos,
                         const int* __restrict__ esrc, const int* __restrict__ etgt,
                         const int* __restrict__ erel, unsigned short* __restrict__ X) {
    int m = blockIdx.x;
    int b = m / (2*E_);
    int r = m - b*2*E_;
    int e = r >> 1, dir = r & 1;
    size_t be = (size_t)b*E_ + e;
    int s = esrc[be], t = etgt[be], rl = erel[be];
    if (dir) { int tmp = s; s = t; t = tmp; rl = -rl; }
    rl = rl < -MAXREL_ ? -MAXREL_ : (rl > MAXREL_ ? MAXREL_ : rl);
    const float* hs = h + ((size_t)(b*L_ + s)) * D_;
    const float* ht = h + ((size_t)(b*L_ + t)) * D_;
    const float* pe = pos + (size_t)(rl + MAXREL_) * D_;
    unsigned short* xr = X + (size_t)m * (3*D_);
    for (int c = threadIdx.x; c < D_; c += blockDim.x) {
        xr[c        ] = f2bf(hs[c]);
        xr[c +   D_ ] = f2bf(ht[c]);
        xr[c + 2*D_ ] = f2bf(pe[c]);
    }
}

// --------------------------------------------------------- bf16 WMMA GEMM ----
#define BM 128
#define BN 64
#define BK 32
#define LDA_S 40     // 32 data + 8 pad ushorts (TDM: 64B interval + 16B pad)
#define LDB_S 40

// Issue a TDM 2D tile load: tile (BK x BM) bf16 from row-major A[Mrows x K]
// into LDS at ldsAddr with 16B padding after every 64B row.
static __device__ __forceinline__ void tdm_load_A(const unsigned short* gptr,
                                                  unsigned ldsAddr,
                                                  int K, int Mrows) {
    unsigned long long ga = (unsigned long long)gptr;
    u32x4 g0;
    g0[0] = 1u;                               // count=1, user mode
    g0[1] = ldsAddr;                          // lds_addr (bytes)
    g0[2] = (unsigned)ga;                     // global_addr[31:0]
    g0[3] = (unsigned)((ga >> 32) & 0x1FFFFFFull) | (2u << 30);  // addr hi + type=2
    i32x8 g1;
    // data_size=1 (2B), pad_enable, pad_interval=3 (16 DW = 64B), pad_amount=3 (4 DW = 16B)
    g1[0] = (int)((1u << 16) | (1u << 20) | (3u << 22) | (3u << 25));
    g1[1] = (int)(((unsigned)K & 0xFFFFu) << 16);          // tensor_dim0 lo16 (bits 63:48)
    g1[2] = (int)((((unsigned)K >> 16) & 0xFFFFu)          // tensor_dim0 hi16
                  | (((unsigned)Mrows & 0xFFFFu) << 16));  // tensor_dim1 lo16
    g1[3] = (int)((((unsigned)Mrows >> 16) & 0xFFFFu)      // tensor_dim1 hi16
                  | ((unsigned)BK << 16));                 // tile_dim0 = 32
    g1[4] = (int)BM;                                       // tile_dim1 = 128, tile_dim2 = 0
    g1[5] = (int)(unsigned)K;                              // tensor_dim0_stride lo32
    g1[6] = 0;                                             // stride0 hi16, stride1 lo16
    g1[7] = 0;                                             // stride1 hi
    i32x4 g2 = {}; i32x4 g3 = {};                          // 2D: groups 2/3 unused
#if defined(__clang_major__) && (__clang_major__ >= 23)
    i32x8 gpad = {};
    __builtin_amdgcn_tensor_load_to_lds(g0, g1, g2, g3, gpad, 0);
#else
    __builtin_amdgcn_tensor_load_to_lds(g0, g1, g2, g3, 0);
#endif
}

__global__ __launch_bounds__(256)
void k_gemm_bf16(const unsigned short* __restrict__ A, const float* __restrict__ W,
                 float* __restrict__ C, int Mrows, int K, int N) {
    __shared__ __attribute__((aligned(16))) unsigned short lA[BM * LDA_S];
    __shared__ __attribute__((aligned(16))) unsigned short lB[BN * LDB_S];
    int tid  = threadIdx.x;
    int wave = tid >> 5, lane = tid & 31;
    int lhalf = lane & 15, hi16 = lane >> 4;
    int mBase = blockIdx.x * BM;
    int nBase = blockIdx.y * BN;
    unsigned ldsA_addr = (unsigned)(unsigned long long)(void*)&lA[0];
    v8f acc[4] = {};

    for (int k0 = 0; k0 < K; k0 += BK) {
        __syncthreads();
        if (wave == 0) {
            // async tensor DMA: A tile (BKxBM bf16) -> LDS, OOB rows -> zero
            tdm_load_A(A + (size_t)mBase*K + k0, ldsA_addr, K, Mrows);
        }
        {   // stage B tile transposed: lB[n][k], f32 -> bf16
            int k = tid >> 3, n0 = (tid & 7) * 8;
            const float* wp = W + (size_t)(k0 + k)*N + nBase + n0;
#pragma unroll
            for (int q = 0; q < 8; ++q)
                lB[(n0 + q)*LDB_S + k] = f2bf(wp[q]);
            if (k0 + BK < K)
                __builtin_prefetch(wp + (size_t)BK*N, 0, 1);  // global_prefetch_b8
        }
        if (wave == 0)
            __builtin_amdgcn_s_wait_tensorcnt(0);             // s_wait_tensorcnt 0
        __syncthreads();

        union Frag { v16bf v; uint4 q[2]; };
        Frag a;
        {   // A fragment: lanes 0-15 -> K[kb..kb+7],K[kb+16..kb+23]
            int m  = (wave << 4) + lhalf;
            int kb = hi16 ? 8 : 0;
            a.q[0] = *(const uint4*)(&lA[m*LDA_S + kb]);
            a.q[1] = *(const uint4*)(&lA[m*LDA_S + kb + 16]);
        }
#pragma unroll
        for (int j = 0; j < 4; ++j) {
            Frag bf;
            int n  = (j << 4) + lhalf;
            int kb = hi16 ? 8 : 0;
            bf.q[0] = *(const uint4*)(&lB[n*LDB_S + kb]);
            bf.q[1] = *(const uint4*)(&lB[n*LDB_S + kb + 16]);
            acc[j] = __builtin_amdgcn_wmma_f32_16x16x32_bf16(
                false, a.v, false, bf.v, (short)0, acc[j], false, false);
        }
    }
    int mw = mBase + (wave << 4) + (hi16 ? 8 : 0);
#pragma unroll
    for (int j = 0; j < 4; ++j) {
        int n = nBase + (j << 4) + lhalf;
#pragma unroll
        for (int r = 0; r < 8; ++r) {
            int m = mw + r;
            if (m < Mrows) C[(size_t)m*N + n] = acc[j][r];
        }
    }
}

// ------------------------------------------------- bias + LN + ReLU -> bf16 ----
__global__ void k_lnrelu(const float* __restrict__ ACC, const float* __restrict__ bias,
                         const float* __restrict__ gam, const float* __restrict__ bet,
                         unsigned short* __restrict__ Y, int N) {
    __shared__ float s1[256], s2[256];
    int m = blockIdx.x, tid = threadIdx.x;
    const float* xr = ACC + (size_t)m*N;
    float sum = 0.f, sq = 0.f;
    for (int n = tid; n < N; n += 256) {
        float x = xr[n] + bias[n]; sum += x; sq += x*x;
    }
    s1[tid] = sum; s2[tid] = sq; __syncthreads();
    for (int s = 128; s; s >>= 1) {
        if (tid < s) { s1[tid] += s1[tid+s]; s2[tid] += s2[tid+s]; }
        __syncthreads();
    }
    float mu  = s1[0] / N;
    float var = s2[0] / N - mu*mu;
    float rs  = rsqrtf(var + 1e-5f);
    unsigned short* yr = Y + (size_t)m*N;
    for (int n = tid; n < N; n += 256) {
        float x = xr[n] + bias[n];
        float y = (x - mu)*rs*gam[n] + bet[n];
        yr[n] = f2bf(y > 0.f ? y : 0.f);
    }
}

// ------------------------------------------------- bias + unit-normalize ----
__global__ void k_normv(const float* __restrict__ ACC, const float* __restrict__ b3,
                        float* __restrict__ V) {
    __shared__ float s1[256];
    int m = blockIdx.x, tid = threadIdx.x;
    const float* xr = ACC + (size_t)m*D_;
    float sq = 0.f;
    for (int n = tid; n < D_; n += 256) { float x = xr[n] + b3[n]; sq += x*x; }
    s1[tid] = sq; __syncthreads();
    for (int s = 128; s; s >>= 1) { if (tid < s) s1[tid] += s1[tid+s]; __syncthreads(); }
    float inv = 1.0f / fmaxf(sqrtf(s1[0]), 1e-8f);
    float* vr = V + (size_t)m*D_;
    for (int n = tid; n < D_; n += 256) vr[n] = (xr[n] + b3[n]) * inv;
}

// ------------------------------------------------------ iteration kernels ----
__global__ void k_zero(float* __restrict__ p, unsigned int n) {
    unsigned int i = blockIdx.x*blockDim.x + threadIdx.x;
    if (i < n) p[i] = 0.0f;
}

__global__ void k_edgegrad(const float* __restrict__ hc, const float* __restrict__ V,
                           const int* __restrict__ esrc, const int* __restrict__ etgt,
                           const float* __restrict__ ew, float* __restrict__ g,
                           const int* __restrict__ dflag) {
    if (*dflag) return;
    __shared__ float r1[128], r2[128];
    int be = blockIdx.x;
    int b  = be / E_;
    float w = ew[be];
    if (w == 0.0f) return;
    int s = esrc[be], t = etgt[be];
    int tid = threadIdx.x;
    const float* hs  = hc + ((size_t)(b*L_ + s)) * D_;
    const float* htp = hc + ((size_t)(b*L_ + t)) * D_;
    const float* vij = V + ((size_t)be*2    ) * D_;
    const float* vji = V + ((size_t)be*2 + 1) * D_;
    float xs[4], xt[4], a[4], c[4];
    float da = 0.f, db = 0.f;
#pragma unroll
    for (int q = 0; q < 4; ++q) {
        int d = tid + q*128;
        xs[q] = hs[d]; xt[q] = htp[d]; a[q] = vij[d]; c[q] = vji[d];
        da += a[q]*xs[q]; db += c[q]*xt[q];
    }
    r1[tid] = da; r2[tid] = db; __syncthreads();
    for (int o = 64; o; o >>= 1) { if (tid < o) { r1[tid]+=r1[tid+o]; r2[tid]+=r2[tid+o]; } __syncthreads(); }
    da = r1[0]; db = r2[0]; __syncthreads();
    float dc = 0.f, dd = 0.f, del[4];
#pragma unroll
    for (int q = 0; q < 4; ++q) {
        float rs = xs[q] - 2.f*a[q]*da;
        float rt = xt[q] - 2.f*c[q]*db;
        del[q] = (rs - rt) * w;
        dc += a[q]*del[q]; dd += c[q]*del[q];
    }
    r1[tid] = dc; r2[tid] = dd; __syncthreads();
    for (int o = 64; o; o >>= 1) { if (tid < o) { r1[tid]+=r1[tid+o]; r2[tid]+=r2[tid+o]; } __syncthreads(); }
    dc = r1[0]; dd = r2[0];
    float* gs = g + ((size_t)(b*L_ + s)) * D_;
    float* gt = g + ((size_t)(b*L_ + t)) * D_;
#pragma unroll
    for (int q = 0; q < 4; ++q) {
        int d = tid + q*128;
        atomicAdd(&gs[d],  del[q] - 2.f*a[q]*dc);
        atomicAdd(&gt[d], -(del[q] - 2.f*c[q]*dd));
    }
}

__global__ void k_update(float* __restrict__ hc, const float* __restrict__ h0,
                         const float* __restrict__ g, const float* __restrict__ eta,
                         const int* __restrict__ dflag) {
    if (*dflag) return;
    __shared__ float r1[128];
    int row = blockIdx.x, tid = threadIdx.x;
    const float denom = (float)E_ * (float)D_ + 1e-8f;
    float*       hr  = hc + (size_t)row*D_;
    const float* h0r = h0 + (size_t)row*D_;
    const float* gr  = g  + (size_t)row*D_;
    float gi[4], sq = 0.f;
#pragma unroll
    for (int q = 0; q < 4; ++q) {
        int d = tid + q*128;
        gi[q] = gr[d]/denom + LAM_*(hr[d] - h0r[d]);
        sq += gi[q]*gi[q];
    }
    r1[tid] = sq; __syncthreads();
    for (int o = 64; o; o >>= 1) { if (tid < o) r1[tid] += r1[tid+o]; __syncthreads(); }
    float n = sqrtf(r1[0]);
    float scale = fminf(MAXGN_ / (n + 1e-8f), 1.0f) * fabsf(*eta);
#pragma unroll
    for (int q = 0; q < 4; ++q) { int d = tid + q*128; hr[d] -= gi[q]*scale; }
}

__global__ void k_energy(const float* __restrict__ hc, const float* __restrict__ V,
                         const int* __restrict__ esrc, const int* __restrict__ etgt,
                         const float* __restrict__ ew, float* __restrict__ curAcc,
                         const int* __restrict__ dflag) {
    if (*dflag) return;
    __shared__ float r1[128], r2[128];
    int be = blockIdx.x;
    int b  = be / E_;
    float w = ew[be];
    if (w == 0.0f) return;
    int s = esrc[be], t = etgt[be];
    int tid = threadIdx.x;
    const float* hs  = hc + ((size_t)(b*L_ + s)) * D_;
    const float* htp = hc + ((size_t)(b*L_ + t)) * D_;
    const float* vij = V + ((size_t)be*2    ) * D_;
    const float* vji = V + ((size_t)be*2 + 1) * D_;
    float xs[4], xt[4], a[4], c[4];
    float da = 0.f, db = 0.f;
#pragma unroll
    for (int q = 0; q < 4; ++q) {
        int d = tid + q*128;
        xs[q] = hs[d]; xt[q] = htp[d]; a[q] = vij[d]; c[q] = vji[d];
        da += a[q]*xs[q]; db += c[q]*xt[q];
    }
    r1[tid] = da; r2[tid] = db; __syncthreads();
    for (int o = 64; o; o >>= 1) { if (tid < o) { r1[tid]+=r1[tid+o]; r2[tid]+=r2[tid+o]; } __syncthreads(); }
    da = r1[0]; db = r2[0]; __syncthreads();
    float el = 0.f;
#pragma unroll
    for (int q = 0; q < 4; ++q) {
        float df = (xs[q] - 2.f*a[q]*da) - (xt[q] - 2.f*c[q]*db);
        el += df*df;
    }
    r1[tid] = el; __syncthreads();
    for (int o = 64; o; o >>= 1) { if (tid < o) r1[tid] += r1[tid+o]; __syncthreads(); }
    if (tid == 0) atomicAdd(curAcc, 0.5f * r1[0] * w);
}

__global__ void k_init(float* slots, int* dflag) { if (threadIdx.x == 0) { slots[0] = 0.f; *dflag = 0; } }
__global__ void k_zero_cur(float* slots)         { if (threadIdx.x == 0) slots[0] = 0.f; }
__global__ void k_set_pre(float* slots) {
    if (threadIdx.x == 0) {
        const float sc = 1.0f / (((float)E_*(float)D_ + 1e-8f) * (float)B_);
        slots[1] = slots[0] * sc;
    }
}
__global__ void k_conv(const float* slots, int* dflag) {
    if (threadIdx.x == 0) {
        const float sc = 1.0f / (((float)E_*(float)D_ + 1e-8f) * (float)B_);
        float cur = slots[0] * sc, pre = slots[1];
        if (fabsf(pre - cur) / (pre + 1e-8f) < EPSC_) *dflag = 1;
    }
}

// ----------------------------------------------------------------- driver ----
extern "C" void kernel_launch(void* const* d_in, const int* in_sizes, int n_in,
                              void* d_out, int out_size, void* d_ws, size_t ws_size,
                              hipStream_t stream) {
    const float* h    = (const float*)d_in[0];
    const float* attn = (const float*)d_in[1];
    const float* eta  = (const float*)d_in[2];
    const float* pos  = (const float*)d_in[3];
    const float* W1 = (const float*)d_in[4];
    const float* b1 = (const float*)d_in[5];
    const float* g1 = (const float*)d_in[6];
    const float* be1= (const float*)d_in[7];
    const float* W2 = (const float*)d_in[8];
    const float* b2 = (const float*)d_in[9];
    const float* g2 = (const float*)d_in[10];
    const float* be2= (const float*)d_in[11];
    const float* W3 = (const float*)d_in[12];
    const float* b3 = (const float*)d_in[13];
    float* out = (float*)d_out;

    char* ws = (char*)d_ws;
    size_t off = 0;
    auto alloc = [&](size_t bytes) -> char* {
        char* p = ws + off;
        off += (bytes + 255) & ~(size_t)255;
        return p;
    };
    float* slots = (float*)alloc(256);
    int*   dflag = (int*)  alloc(256);
    int*   esrc  = (int*)  alloc((size_t)B_*E_*sizeof(int));
    int*   etgt  = (int*)  alloc((size_t)B_*E_*sizeof(int));
    int*   erel  = (int*)  alloc((size_t)B_*E_*sizeof(int));
    float* ew    = (float*)alloc((size_t)B_*E_*sizeof(float));
    float* vals  = (float*)alloc((size_t)B_*L_*TOPK_*sizeof(float));
    int*   idxs  = (int*)  alloc((size_t)B_*L_*TOPK_*sizeof(int));
    float* hc    = (float*)alloc((size_t)B_*L_*D_*sizeof(float));
    float* gbuf  = (float*)alloc((size_t)B_*L_*D_*sizeof(float));
    unsigned short* X = (unsigned short*)alloc((size_t)M_*(3*D_)*sizeof(unsigned short));
    float* V     = (float*)X;   // V (M_*D_ f32) reuses X region after GEMM1
    unsigned short* Y = (unsigned short*)alloc((size_t)M_*HID_*sizeof(unsigned short));
    float* ACC   = (float*)alloc((size_t)M_*HID_*sizeof(float));

    // ---- phase A: graph construction ----
    k_topk <<<B_*L_, 32, 0, stream>>>(attn, vals, idxs);
    k_colsm<<<B_*TOPK_, 256, 0, stream>>>(vals, idxs, esrc, etgt, erel, ew);
    k_chain<<<(B_*CHAIN_ + 255)/256, 256, 0, stream>>>(attn, esrc, etgt, erel, ew);

    // ---- phase B: edge MLP via bf16 WMMA + TDM ----
    k_buildX<<<M_, 256, 0, stream>>>(h, pos, esrc, etgt, erel, X);
    dim3 g1d((M_ + BM - 1)/BM, HID_/BN);
    k_gemm_bf16<<<g1d, 256, 0, stream>>>(X, W1, ACC, M_, 3*D_, HID_);
    k_lnrelu  <<<M_, 256, 0, stream>>>(ACC, b1, g1, be1, Y, HID_);
    dim3 g2d((M_ + BM - 1)/BM, HID_/BN);
    k_gemm_bf16<<<g2d, 256, 0, stream>>>(Y, W2, ACC, M_, HID_, HID_);
    k_lnrelu  <<<M_, 256, 0, stream>>>(ACC, b2, g2, be2, Y, HID_);
    dim3 g3d((M_ + BM - 1)/BM, D_/BN);
    k_gemm_bf16<<<g3d, 256, 0, stream>>>(Y, W3, ACC, M_, HID_, D_);
    k_normv   <<<M_, 256, 0, stream>>>(ACC, b3, V);

    // ---- phase C: iterative refinement ----
    (void)hipMemcpyAsync(hc, h, (size_t)B_*L_*D_*sizeof(float),
                         hipMemcpyDeviceToDevice, stream);
    k_init<<<1, 32, 0, stream>>>(slots, dflag);
    k_energy<<<B_*E_, 128, 0, stream>>>(hc, V, esrc, etgt, ew, slots, dflag);
    k_set_pre<<<1, 32, 0, stream>>>(slots);

    const unsigned int nElems = (unsigned int)(B_*L_*D_);
    for (int step = 0; step < KSTEPS_; ++step) {
        k_zero    <<<(nElems + 255)/256, 256, 0, stream>>>(gbuf, nElems);
        k_edgegrad<<<B_*E_, 128, 0, stream>>>(hc, V, esrc, etgt, ew, gbuf, dflag);
        k_update  <<<B_*L_, 128, 0, stream>>>(hc, h, gbuf, eta, dflag);
        k_zero_cur<<<1, 32, 0, stream>>>(slots);
        k_energy  <<<B_*E_, 128, 0, stream>>>(hc, V, esrc, etgt, ew, slots, dflag);
        k_conv    <<<1, 32, 0, stream>>>(slots, dflag);
    }

    (void)hipMemcpyAsync(out, hc, (size_t)B_*L_*D_*sizeof(float),
                         hipMemcpyDeviceToDevice, stream);
}
